// SemanticFeatureCorrector_74526272520690
// MI455X (gfx1250) — matrix operand order repaired and checked by reading, not compiled, not run on hardware
//
#include <hip/hip_runtime.h>

// ---------------------------------------------------------------------------
// Types for CDNA5 WMMA (wave32, bf16 in / f32 accumulate)
// ---------------------------------------------------------------------------
typedef __bf16 bf16_t;
typedef __attribute__((ext_vector_type(16))) __bf16 v16bf;
typedef __attribute__((ext_vector_type(8)))  __bf16 v8bf;
typedef __attribute__((ext_vector_type(8)))  float   v8f;

#define D_DIM 2048
#define C_DIM 80
#define B_DIM 64
#define HW    196
#define NPAD  208   // HW padded up to a multiple of 16 for WMMA tiling

static __device__ __forceinline__ v8f wmma_bf16(v16bf a, v16bf b, v8f c) {
    // D = A(16x32 bf16) * B(32x16 bf16) + C(16x16 f32)
    return __builtin_amdgcn_wmma_f32_16x16x32_bf16(
        /*neg_a=*/false, a, /*neg_b=*/false, b,
        /*c_mod=*/(short)0, c, /*reuse_a=*/false, /*reuse_b=*/false);
}

// A-matrix fragment (16x32 bf16, contiguous K order used by this code, not the ISA table order): lane holds row M = lane&15.
// lanes 0-15 : VGPR0-3 -> K=0..7,   VGPR4-7 -> K=8..15
// lanes 16-31: VGPR0-3 -> K=16..23, VGPR4-7 -> K=24..31
static __device__ __forceinline__ v16bf load_a_frag(const bf16_t* __restrict__ rowptr,
                                                    int k0, int khalf) {
    v8bf lo = *(const v8bf*)(rowptr + k0 + khalf * 16);
    v8bf hi = *(const v8bf*)(rowptr + k0 + 8 + khalf * 16);
    v16bf a;
#pragma unroll
    for (int i = 0; i < 8; ++i) { a[i] = lo[i]; a[i + 8] = hi[i]; }
    return a;
}

// B-matrix fragment: lane holds column N = lane&15; 16 contiguous K values
// per half-wave. Built from two 16-B loads so a 16-B-aligned (padded) LDS row
// stride is legal.
static __device__ __forceinline__ v16bf load_b_frag(const bf16_t* __restrict__ p) {
    v8bf lo = *(const v8bf*)(p);
    v8bf hi = *(const v8bf*)(p + 8);
    v16bf b;
#pragma unroll
    for (int i = 0; i < 8; ++i) { b[i] = lo[i]; b[i + 8] = hi[i]; }
    return b;
}

static __device__ __forceinline__ v8f zero_v8f() {
    v8f z = {0.f, 0.f, 0.f, 0.f, 0.f, 0.f, 0.f, 0.f};
    return z;
}

// Async global(16B/lane) -> LDS copy, tracked by ASYNCcnt.
static __device__ __forceinline__ void async_copy_b128(unsigned lds_off, const void* gptr) {
    asm volatile("global_load_async_to_lds_b128 %0, %1, off"
                 :: "v"(lds_off), "v"((unsigned long long)(uintptr_t)gptr)
                 : "memory");
}

// ---------------------------------------------------------------------------
// fp32 -> bf16 conversion
// ---------------------------------------------------------------------------
__global__ __launch_bounds__(256) void f32_to_bf16_kernel(const float* __restrict__ src,
                                                          bf16_t* __restrict__ dst, int n) {
    int i = blockIdx.x * blockDim.x + threadIdx.x;
    if (i < n) dst[i] = (bf16_t)src[i];
}

// Wa_w [196][2048] -> bf16 [208][2048], zero-padded rows 196..207
__global__ __launch_bounds__(256) void wa_pad_bf16_kernel(const float* __restrict__ src,
                                                          bf16_t* __restrict__ dst) {
    int i = blockIdx.x * blockDim.x + threadIdx.x;
    if (i >= NPAD * D_DIM) return;
    int r = i / D_DIM;
    dst[i] = (r < HW) ? (bf16_t)src[i] : (bf16_t)0.f;
}

// ---------------------------------------------------------------------------
// GEMM 1+2 (LDS-pipelined): Q = sem*Wq^T + bq ; K = sem*Wk^T + bk  (z selects)
// M = 5120, N = 2048, K = 2048.
// Block = 256 threads (8 waves), macro tile 128(M) x 128(N).
// K staged in 64-wide panels, double-buffered in LDS via async copies.
// LDS row stride padded 64 -> 72 bf16 (144 B = 36 banks): 36*r mod 64 is
// distinct for r in [0,16) -> conflict-free 16-row fragment reads.
// Each wave computes a 32x64 register tile: 2 M-subtiles x 4 N-subtiles,
// 8 WMMAs per 32-K substep (A-frag shared over N, B-frag shared over M).
// ---------------------------------------------------------------------------
#define KTILE     64
#define KSTAGES   (D_DIM / KTILE)          // 32
#define LDS_STR   72                       // padded row stride in bf16 elements
#define PANEL_EL  (128 * LDS_STR)          // 9216 bf16 per buffer
#define PANEL_B   (PANEL_EL * 2)           // 18432 bytes per buffer

__global__ __launch_bounds__(256) void gemm_qk_lds_kernel(const bf16_t* __restrict__ semb,
                                                          const bf16_t* __restrict__ wq,
                                                          const bf16_t* __restrict__ wk,
                                                          const float* __restrict__ bq,
                                                          const float* __restrict__ bk,
                                                          bf16_t* __restrict__ qout,
                                                          bf16_t* __restrict__ kout) {
    const bf16_t* W    = blockIdx.z ? wk : wq;
    const float*  bias = blockIdx.z ? bk : bq;
    bf16_t*       out  = blockIdx.z ? kout : qout;

    __shared__ __align__(32) bf16_t sm[4 * PANEL_EL];   // [A0|A1|B0|B1] = 72 KB

    const int tid   = threadIdx.x;
    const int lane  = tid & 31;
    const int wave  = tid >> 5;            // 0..7
    const int khalf = lane >> 4;
    const int nlane = lane & 15;
    const int m_wv  = (wave >> 1) * 32;    // 0..96
    const int n_wv  = (wave & 1) * 64;     // 0 or 64

    const int mbase = blockIdx.x * 128;
    const int nbase = blockIdx.y * 128;

    const unsigned ldsA = (unsigned)(uintptr_t)(void*)&sm[0];
    const unsigned ldsB = ldsA + 2 * PANEL_B;

    // Stage issue: 1024 16-byte chunks per panel; 256 threads x 4 rounds.
    auto issue_stage = [&](int s) {
        const int k0  = s * KTILE;
        const int buf = s & 1;
#pragma unroll
        for (int r = 0; r < 4; ++r) {
            const int c   = r * 256 + tid;   // 0..1023
            const int row = c >> 3;          // 0..127
            const int col = c & 7;           // 16B chunk within 128B of data
            const unsigned loff = buf * PANEL_B + row * (LDS_STR * 2) + col * 16;
            async_copy_b128(ldsA + loff,
                            semb + (size_t)(mbase + row) * D_DIM + k0 + col * 8);
            async_copy_b128(ldsB + loff,
                            W + (size_t)(nbase + row) * D_DIM + k0 + col * 8);
        }
    };

    v8f acc[2][4];
#pragma unroll
    for (int mt = 0; mt < 2; ++mt)
#pragma unroll
        for (int t = 0; t < 4; ++t) acc[mt][t] = zero_v8f();

    issue_stage(0);

    for (int s = 0; s < KSTAGES; ++s) {
        if (s + 1 < KSTAGES) {
            issue_stage(s + 1);
            // 8 newest ops remain -> everything for stage s has landed
            asm volatile("s_wait_asynccnt 0x8" ::: "memory");
        } else {
            asm volatile("s_wait_asynccnt 0x0" ::: "memory");
        }
        __syncthreads();

        const int buf = s & 1;
        const bf16_t* Abuf = sm + buf * PANEL_EL;
        const bf16_t* Bbuf = sm + 2 * PANEL_EL + buf * PANEL_EL;

#pragma unroll
        for (int ks = 0; ks < KTILE / 32; ++ks) {
            // Hoist all fragment loads of this substep before the WMMAs so the
            // scheduler can overlap ds_load latency with XDL work.
            v16bf afr[2], bfr[4];
#pragma unroll
            for (int mt = 0; mt < 2; ++mt) {
                const bf16_t* arow = Abuf + (size_t)(m_wv + mt * 16 + nlane) * LDS_STR;
                afr[mt] = load_a_frag(arow, ks * 32, khalf);
            }
#pragma unroll
            for (int t = 0; t < 4; ++t) {
                const bf16_t* brow = Bbuf + (size_t)(n_wv + t * 16 + nlane) * LDS_STR +
                                     ks * 32 + khalf * 16;
                bfr[t] = load_b_frag(brow);
            }
#pragma unroll
            for (int t = 0; t < 4; ++t)
#pragma unroll
                for (int mt = 0; mt < 2; ++mt)
                    acc[mt][t] = wmma_bf16(afr[mt], bfr[t], acc[mt][t]);
        }
        __syncthreads();   // protect buffer reuse two stages later
    }

    // Epilogue: bias add + bf16 store
#pragma unroll
    for (int t = 0; t < 4; ++t) {
        const int n = nbase + n_wv + t * 16 + nlane;
        const float bv = bias[n];
#pragma unroll
        for (int mt = 0; mt < 2; ++mt) {
#pragma unroll
            for (int r = 0; r < 8; ++r) {
                const int m = mbase + m_wv + mt * 16 + r + khalf * 8;
                out[(size_t)m * D_DIM + n] = (bf16_t)(acc[mt][t][r] + bv);
            }
        }
    }
}

// ---------------------------------------------------------------------------
// GEMM 3: S_raw[b] = Q_b (80x2048) * K_b^T (2048x80) / sqrt(D)
// One wave per 16x16 tile; grid = (5, 5, 64).
// ---------------------------------------------------------------------------
__global__ __launch_bounds__(32) void gemm_s_kernel(const bf16_t* __restrict__ Qb,
                                                    const bf16_t* __restrict__ Kb,
                                                    float* __restrict__ S) {
    const int lane  = threadIdx.x;
    const int khalf = lane >> 4;
    const int nlane = lane & 15;
    const int bidx  = blockIdx.z;
    const int m0    = blockIdx.x * 16;
    const int n0    = blockIdx.y * 16;

    const bf16_t* rowA = Qb + ((size_t)bidx * C_DIM + m0 + nlane) * D_DIM;
    const bf16_t* rowB = Kb + ((size_t)bidx * C_DIM + n0 + nlane) * D_DIM;

    v8f acc = zero_v8f();
    for (int k0 = 0; k0 < D_DIM; k0 += 32) {
        v16bf a = load_a_frag(rowA, k0, khalf);
        v16bf b = load_b_frag(rowB + k0 + khalf * 16);
        acc = wmma_bf16(a, b, acc);
    }

    const float scale = 0.022097086912079608f;  // 1/sqrt(2048)
#pragma unroll
    for (int r = 0; r < 8; ++r) {
        const int m = m0 + r + khalf * 8;
        S[((size_t)bidx * C_DIM + m) * C_DIM + n0 + nlane] = acc[r] * scale;
    }
}

// ---------------------------------------------------------------------------
// Masked softmax over rows of S (length 80). One wave32 per row.
// ---------------------------------------------------------------------------
__global__ __launch_bounds__(32) void softmax_mask_kernel(float* __restrict__ S,
                                                          const float* __restrict__ mask) {
    const int row  = blockIdx.x;           // b*80 + c
    const int c    = row % C_DIM;
    const int lane = threadIdx.x;
    float*       Srow = S + (size_t)row * C_DIM;
    const float* mrow = mask + (size_t)c * C_DIM;

    float v[3];
    int cnt = 0;
    float mx = -3.402823e38f;
    for (int j = lane; j < C_DIM; j += 32) {
        float x = Srow[j] * mrow[j];
        v[cnt++] = x;
        mx = fmaxf(mx, x);
    }
#pragma unroll
    for (int o = 16; o > 0; o >>= 1) mx = fmaxf(mx, __shfl_xor(mx, o, 32));

    float s = 0.f;
    cnt = 0;
    for (int j = lane; j < C_DIM; j += 32) {
        float e = __expf(v[cnt] - mx);
        v[cnt++] = e;
        s += e;
    }
#pragma unroll
    for (int o = 16; o > 0; o >>= 1) s += __shfl_xor(s, o, 32);

    const float inv = 1.0f / s;
    cnt = 0;
    for (int j = lane; j < C_DIM; j += 32) Srow[j] = v[cnt++] * inv;
}

// ---------------------------------------------------------------------------
// GEMM 4: adjust = sigmoid(sem * Wa^T + ba).  M = 5120, N = 208(pad), K = 2048.
// Block = 4 waves; each wave owns one 16x16 N-tile (13 valid tiles).
// ---------------------------------------------------------------------------
__global__ __launch_bounds__(128) void gemm_adjust_kernel(const bf16_t* __restrict__ semb,
                                                          const bf16_t* __restrict__ wab,
                                                          const float* __restrict__ ab,
                                                          float* __restrict__ adjust) {
    const int lane  = threadIdx.x & 31;
    const int wave  = threadIdx.x >> 5;
    const int khalf = lane >> 4;
    const int nlane = lane & 15;
    const int m0    = blockIdx.x * 16;
    const int ntile = blockIdx.y * 4 + wave;
    if (ntile >= NPAD / 16) return;
    const int n0 = ntile * 16;

    const bf16_t* rowA = semb + (size_t)(m0 + nlane) * D_DIM;
    const bf16_t* rowB = wab + (size_t)(n0 + nlane) * D_DIM;

    v8f acc = zero_v8f();
    for (int k0 = 0; k0 < D_DIM; k0 += 32) {
        v16bf a = load_a_frag(rowA, k0, khalf);
        v16bf b = load_b_frag(rowB + k0 + khalf * 16);
        acc = wmma_bf16(a, b, acc);
    }

    const int n = n0 + nlane;
    if (n < HW) {
        const float bias = ab[n];
#pragma unroll
        for (int r = 0; r < 8; ++r) {
            const int m = m0 + r + khalf * 8;
            const float x = acc[r] + bias;
            adjust[(size_t)m * HW + n] = 1.0f / (1.0f + __expf(-x));
        }
    }
}

// ---------------------------------------------------------------------------
// Final fuse: out = (features + S @ features_flat) * adjust
// ---------------------------------------------------------------------------
__global__ __launch_bounds__(256) void fuse_out_kernel(const float* __restrict__ feat,
                                                       const float* __restrict__ S,
                                                       const float* __restrict__ adjust,
                                                       float* __restrict__ out, int total) {
    const int idx = blockIdx.x * blockDim.x + threadIdx.x;
    if (idx >= total) return;
    const int h  = idx % HW;
    const int bc = idx / HW;        // b*80 + c
    const int b  = bc / C_DIM;

    const float* Srow = S + (size_t)bc * C_DIM;
    const float* fb   = feat + (size_t)b * C_DIM * HW + h;
    float ctx = 0.f;
#pragma unroll 4
    for (int k = 0; k < C_DIM; ++k) ctx += Srow[k] * fb[(size_t)k * HW];
    out[idx] = (feat[idx] + ctx) * adjust[idx];
}

// ---------------------------------------------------------------------------
// Workspace layout (all 256-B aligned)
// ---------------------------------------------------------------------------
static constexpr size_t OFF_SEMB = 0;                                                // bf16 sem     20,971,520 B
static constexpr size_t OFF_WQB  = OFF_SEMB + (size_t)B_DIM * C_DIM * D_DIM * 2;     // bf16 Wq       8,388,608 B
static constexpr size_t OFF_WKB  = OFF_WQB + (size_t)D_DIM * D_DIM * 2;              // bf16 Wk       8,388,608 B
static constexpr size_t OFF_WAB  = OFF_WKB + (size_t)D_DIM * D_DIM * 2;              // bf16 Wa pad     851,968 B
static constexpr size_t OFF_QB   = OFF_WAB + (size_t)NPAD * D_DIM * 2;               // bf16 Q       20,971,520 B
static constexpr size_t OFF_KB   = OFF_QB + (size_t)B_DIM * C_DIM * D_DIM * 2;       // bf16 K       20,971,520 B
static constexpr size_t OFF_S    = OFF_KB + (size_t)B_DIM * C_DIM * D_DIM * 2;       // f32  S        1,638,400 B
static constexpr size_t OFF_ADJ  = OFF_S + (size_t)B_DIM * C_DIM * C_DIM * 4;        // f32  adjust   4,014,080 B

extern "C" void kernel_launch(void* const* d_in, const int* in_sizes, int n_in,
                              void* d_out, int out_size, void* d_ws, size_t ws_size,
                              hipStream_t stream) {
    (void)in_sizes; (void)n_in; (void)out_size; (void)ws_size;

    const float* features = (const float*)d_in[0];  // [64,80,14,14]
    const float* sem      = (const float*)d_in[1];  // [64,80,2048]
    const float* mask     = (const float*)d_in[2];  // [80,80]
    const float* Wq_w     = (const float*)d_in[3];  // [2048,2048]
    const float* Wq_b     = (const float*)d_in[4];  // [2048]
    const float* Wk_w     = (const float*)d_in[5];  // [2048,2048]
    const float* Wk_b     = (const float*)d_in[6];  // [2048]
    const float* Wa_w     = (const float*)d_in[7];  // [196,2048]
    const float* Wa_b     = (const float*)d_in[8];  // [196]
    float* out = (float*)d_out;

    char* ws = (char*)d_ws;
    bf16_t* semb = (bf16_t*)(ws + OFF_SEMB);
    bf16_t* wqb  = (bf16_t*)(ws + OFF_WQB);
    bf16_t* wkb  = (bf16_t*)(ws + OFF_WKB);
    bf16_t* wab  = (bf16_t*)(ws + OFF_WAB);
    bf16_t* Qb   = (bf16_t*)(ws + OFF_QB);
    bf16_t* Kb   = (bf16_t*)(ws + OFF_KB);
    float*  Sbuf = (float*)(ws + OFF_S);
    float*  adjb = (float*)(ws + OFF_ADJ);

    // 1. bf16 conversions
    const int nsem = B_DIM * C_DIM * D_DIM;
    f32_to_bf16_kernel<<<(nsem + 255) / 256, 256, 0, stream>>>(sem, semb, nsem);
    const int nw = D_DIM * D_DIM;
    f32_to_bf16_kernel<<<(nw + 255) / 256, 256, 0, stream>>>(Wq_w, wqb, nw);
    f32_to_bf16_kernel<<<(nw + 255) / 256, 256, 0, stream>>>(Wk_w, wkb, nw);
    wa_pad_bf16_kernel<<<(NPAD * D_DIM + 255) / 256, 256, 0, stream>>>(Wa_w, wab);

    // 2. Q and K projections (z selects): 128x128 macro tiles, LDS async pipeline
    gemm_qk_lds_kernel<<<dim3((B_DIM * C_DIM) / 128, D_DIM / 128, 2), 256, 0, stream>>>(
        semb, wqb, wkb, Wq_b, Wk_b, Qb, Kb);

    // 3. batched scores S_raw = Q K^T / sqrt(D)
    gemm_s_kernel<<<dim3(C_DIM / 16, C_DIM / 16, B_DIM), 32, 0, stream>>>(Qb, Kb, Sbuf);

    // 4. masked softmax over rows of 80
    softmax_mask_kernel<<<B_DIM * C_DIM, 32, 0, stream>>>(Sbuf, mask);

    // 5. sigmoid gate GEMM
    gemm_adjust_kernel<<<dim3((B_DIM * C_DIM) / 16, (NPAD / 16 + 3) / 4, 1), 128, 0, stream>>>(
        semb, wab, Wa_b, adjb);

    // 6. context + residual + gate
    const int total = B_DIM * C_DIM * HW;
    fuse_out_kernel<<<(total + 255) / 256, 256, 0, stream>>>(features, Sbuf, adjb, out, total);
}
